// FAVORPlusAttention_61632780697954
// MI455X (gfx1250) — compile-verified, matrix operand order checked by reading, not run on hardware
//
#include <hip/hip_runtime.h>
#include <hip/hip_bf16.h>

typedef __bf16 bf16_t;
typedef __attribute__((ext_vector_type(16))) __bf16 v16bf;
typedef __attribute__((ext_vector_type(8)))  float  v8f;
typedef __attribute__((ext_vector_type(4)))  unsigned int v4u;

#define HN 8
#define KDIM 64
#define RR 256
#define SS 4096
#define BBATCH 4
#define DD 512
#define CC 128
#define NC 32          // chunks per sequence
#define NBH 32         // B*H
#define EPSF 1e-3f
#define SCALEF 0.125f  // 1/sqrt(64)

__device__ __forceinline__ bf16_t f2bf(float f) {
  union { float f; unsigned int u; } x; x.f = f;
  unsigned int r = x.u + 0x7fffu + ((x.u >> 16) & 1u);  // round-to-nearest-even
  unsigned short h = (unsigned short)(r >> 16);
  bf16_t o; __builtin_memcpy(&o, &h, 2); return o;
}

// Load one 16x32 bf16 WMMA operand fragment.
// Per ISA 7.12.2: lane (row/col = lane&15) holds K in [kb,kb+8) and [kb+16,kb+24)
// where kb = (lane>=16)?8:0 relative to the 32-wide K step. Caller passes
// p = base + row*ld + kstep + kboff; we grab two 16-byte chunks.
__device__ __forceinline__ v16bf load_frag(const bf16_t* p) {
  union { v16bf v; v4u u[2]; } f;
  f.u[0] = *(const v4u*)p;
  f.u[1] = *(const v4u*)(p + 16);
  return f.v;
}

// ---------------------------------------------------------------------------
// Stage 0: fp32 -> bf16 conversions / weight transposes
// ---------------------------------------------------------------------------
__global__ void cvt_f32_bf16(const float* __restrict__ in, bf16_t* __restrict__ out, int n) {
  int i = blockIdx.x * 256 + threadIdx.x;
  if (i < n) out[i] = f2bf(in[i]);
}

__global__ void transpose_cvt(const float* __restrict__ W, bf16_t* __restrict__ Wt,
                              int R, int C) {
  int i = blockIdx.x * 256 + threadIdx.x;
  if (i < R * C) {
    int r = i / C, c = i - r * C;
    Wt[(size_t)c * R + r] = f2bf(W[(size_t)r * C + c]);
  }
}

// ---------------------------------------------------------------------------
// GEMM: C[M,512] = A[M,512] @ Bt[512,512]^T  (A,Bt bf16 row-major)
// OUTMODE 0: bf16 C; OUTMODE 1: fp32 C + bias
// Block 256 thr = 8 waves; block tile 128x128; wave tile 32x64 (2x4 WMMA tiles)
// ---------------------------------------------------------------------------
template <int OUTMODE>
__global__ __launch_bounds__(256) void gemm_bf16_kernel(
    const bf16_t* __restrict__ A, const bf16_t* __restrict__ Bt,
    void* __restrict__ Cout, const float* __restrict__ bias) {
  constexpr int K = DD, N = DD;
  const int lane = threadIdx.x & 31, w = threadIdx.x >> 5;
  const int rA = lane & 15, hlf = lane >> 4, kboff = hlf * 8, cN = rA;
  const int mBase = blockIdx.y * 128 + (w & 3) * 32;
  const int nBase = blockIdx.x * 128 + (w >> 2) * 64;

  v8f acc[2][4];
#pragma unroll
  for (int i = 0; i < 2; ++i)
#pragma unroll
    for (int j = 0; j < 4; ++j) acc[i][j] = {};

  for (int k0 = 0; k0 < K; k0 += 32) {
    v16bf af[2], bf[4];
#pragma unroll
    for (int mt = 0; mt < 2; ++mt)
      af[mt] = load_frag(A + (size_t)(mBase + mt * 16 + rA) * K + k0 + kboff);
#pragma unroll
    for (int nt = 0; nt < 4; ++nt)
      bf[nt] = load_frag(Bt + (size_t)(nBase + nt * 16 + rA) * K + k0 + kboff);
#pragma unroll
    for (int mt = 0; mt < 2; ++mt)
#pragma unroll
      for (int nt = 0; nt < 4; ++nt)
        acc[mt][nt] = __builtin_amdgcn_wmma_f32_16x16x32_bf16(
            false, af[mt], false, bf[nt], (short)0, acc[mt][nt], false, false);
  }
#pragma unroll
  for (int mt = 0; mt < 2; ++mt)
#pragma unroll
    for (int nt = 0; nt < 4; ++nt)
#pragma unroll
      for (int vi = 0; vi < 8; ++vi) {
        int row = mBase + mt * 16 + vi + 8 * hlf;
        int col = nBase + nt * 16 + cN;
        float val = acc[mt][nt][vi];
        if (OUTMODE == 1) ((float*)Cout)[(size_t)row * N + col] = val + bias[col];
        else              ((bf16_t*)Cout)[(size_t)row * N + col] = f2bf(val);
      }
}

// ---------------------------------------------------------------------------
// Features: F[(bs,h), r] = exp(SCALE*(q·rf_r - rowmax)) + EPS
// A = q rows (64 wide), Bt = rf [256,64]. Block: 128 feature rows, wave: 16.
// B-fragments loaded in groups of 4 N-tiles so 16 b128 loads are in flight
// before the first consuming WMMA (avoids s_wait_loadcnt 0 serialization).
// ---------------------------------------------------------------------------
__global__ __launch_bounds__(256) void feature_kernel(
    const bf16_t* __restrict__ Q, const bf16_t* __restrict__ rfb,
    bf16_t* __restrict__ F) {
  const int lane = threadIdx.x & 31, w = threadIdx.x >> 5;
  const int rA = lane & 15, hlf = lane >> 4, kboff = hlf * 8, cN = rA;
  const int rowBase = blockIdx.x * 128 + w * 16;

  v16bf af[2];
  {
    int fr = rowBase + rA;
    int bs = fr >> 3, h = fr & 7;
    const bf16_t* p = Q + (size_t)bs * DD + h * KDIM + kboff;
    af[0] = load_frag(p);
    af[1] = load_frag(p + 32);
  }
  v8f acc[16];
#pragma unroll
  for (int ntg = 0; ntg < 4; ++ntg) {
    v16bf bfr[8];
#pragma unroll
    for (int i = 0; i < 4; ++i) {
      const bf16_t* p = rfb + (size_t)((ntg * 4 + i) * 16 + rA) * KDIM + kboff;
      bfr[2 * i]     = load_frag(p);
      bfr[2 * i + 1] = load_frag(p + 32);
    }
#pragma unroll
    for (int i = 0; i < 4; ++i) {
      int nt = ntg * 4 + i;
      acc[nt] = {};
      acc[nt] = __builtin_amdgcn_wmma_f32_16x16x32_bf16(
          false, af[0], false, bfr[2 * i], (short)0, acc[nt], false, false);
      acc[nt] = __builtin_amdgcn_wmma_f32_16x16x32_bf16(
          false, af[1], false, bfr[2 * i + 1], (short)0, acc[nt], false, false);
    }
  }
  float mrow[8];
#pragma unroll
  for (int vi = 0; vi < 8; ++vi) {
    float m = -3.0e38f;
#pragma unroll
    for (int nt = 0; nt < 16; ++nt) m = fmaxf(m, acc[nt][vi]);
#pragma unroll
    for (int off = 1; off < 16; off <<= 1) m = fmaxf(m, __shfl_xor(m, off, 16));
    mrow[vi] = m;
  }
#pragma unroll
  for (int nt = 0; nt < 16; ++nt)
#pragma unroll
    for (int vi = 0; vi < 8; ++vi) {
      int fr = rowBase + vi + 8 * hlf;
      float val = __expf(SCALEF * (acc[nt][vi] - mrow[vi])) + EPSF;
      F[(size_t)fr * RR + nt * 16 + cN] = f2bf(val);
    }
}

// ---------------------------------------------------------------------------
// Pass A: per-chunk extended state sum  Ssum[bh,chunk][80,256] = vExt @ kf
// vExt: rows 0..63 = v^T, row 64 = ones (-> Z colsum), rows 65..79 = 0
// ---------------------------------------------------------------------------
__global__ __launch_bounds__(256) void chunk_sum_kernel(
    const bf16_t* __restrict__ KF, const bf16_t* __restrict__ V,
    float* __restrict__ Ssum) {
  __shared__ bf16_t kfT[RR * CC];    // [r][c]  64 KB
  __shared__ bf16_t vExt[80 * CC];   // [d][c]  20 KB
  const int tid = threadIdx.x;
  const int chunk = blockIdx.x, bh = blockIdx.y;
  const int b = bh >> 3, h = bh & 7;
  const int s0 = chunk * CC;

  for (int idx = tid; idx < CC * RR; idx += 256) {
    int c = idx >> 8, r = idx & 255;
    size_t fr = (size_t)(b * SS + s0 + c) * HN + h;
    kfT[r * CC + c] = KF[fr * RR + r];
  }
  for (int idx = tid; idx < 80 * CC; idx += 256) {
    int d = idx >> 7, c = idx & 127;
    bf16_t val;
    if (d < KDIM)       val = V[(size_t)(b * SS + s0 + c) * DD + h * KDIM + d];
    else if (d == KDIM) val = f2bf(1.0f);
    else                val = f2bf(0.0f);
    vExt[idx] = val;
  }
  __syncthreads();

  const int lane = tid & 31, w = tid >> 5;
  const int rA = lane & 15, hlf = lane >> 4, kboff = hlf * 8, cN = rA;
  float* out = Ssum + ((size_t)bh * NC + chunk) * (80 * RR);

  for (int t = w; t < 80; t += 8) {           // 5 M-tiles x 16 N-tiles
    int mt = t >> 4, nt = t & 15;
    v16bf afr[4], bfr[4];
#pragma unroll
    for (int ks = 0; ks < 4; ++ks) {          // K = C = 128
      afr[ks] = load_frag(vExt + (mt * 16 + rA) * CC + ks * 32 + kboff);
      bfr[ks] = load_frag(kfT  + (nt * 16 + rA) * CC + ks * 32 + kboff);
    }
    v8f acc = {};
#pragma unroll
    for (int ks = 0; ks < 4; ++ks)
      acc = __builtin_amdgcn_wmma_f32_16x16x32_bf16(
          false, afr[ks], false, bfr[ks], (short)0, acc, false, false);
#pragma unroll
    for (int vi = 0; vi < 8; ++vi)
      out[(size_t)(mt * 16 + vi + 8 * hlf) * RR + nt * 16 + cN] = acc[vi];
  }
}

// ---------------------------------------------------------------------------
// Pass B: in-place exclusive prefix over the 32 chunks (per bh, per element)
// ---------------------------------------------------------------------------
__global__ __launch_bounds__(256) void prefix_kernel(float* __restrict__ S) {
  int gid = blockIdx.x * 256 + threadIdx.x;          // 32 * 20480 total
  int bh = gid / (80 * RR);
  int e  = gid - bh * (80 * RR);
  float* p = S + (size_t)bh * NC * (80 * RR) + e;
  float run = 0.f;
  for (int c = 0; c < NC; ++c) {
    float t = p[(size_t)c * (80 * RR)];
    p[(size_t)c * (80 * RR)] = run;
    run += t;
  }
}

// ---------------------------------------------------------------------------
// Pass C: per-chunk output.
//   A    = tril(qf @ kf^T)                       (128x128, K=256)
//   numE = qf @ StB^T + A @ vExt^T               (128x80, cols 64 = den part)
//   out  = num / (den + EPS)  -> Attn bf16 [B*S, 512]
// ---------------------------------------------------------------------------
__global__ __launch_bounds__(256) void favor_kernel(
    const bf16_t* __restrict__ QF, const bf16_t* __restrict__ KF,
    const bf16_t* __restrict__ V,  const float* __restrict__ Spre,
    bf16_t* __restrict__ Attn) {
  __shared__ bf16_t StB[80 * RR];    // prefix state (bf16)  40 KB
  __shared__ bf16_t vExt[80 * CC];   //                      20 KB
  __shared__ bf16_t Abuf[CC * CC];   // masked A (bf16)      32 KB
  const int tid = threadIdx.x;
  const int chunk = blockIdx.x, bh = blockIdx.y;
  const int b = bh >> 3, h = bh & 7;
  const int s0 = chunk * CC;

  {
    const float* src = Spre + ((size_t)bh * NC + chunk) * (80 * RR);
    for (int idx = tid; idx < 80 * RR; idx += 256) StB[idx] = f2bf(src[idx]);
    for (int idx = tid; idx < 80 * CC; idx += 256) {
      int d = idx >> 7, c = idx & 127;
      bf16_t val;
      if (d < KDIM)       val = V[(size_t)(b * SS + s0 + c) * DD + h * KDIM + d];
      else if (d == KDIM) val = f2bf(1.0f);
      else                val = f2bf(0.0f);
      vExt[idx] = val;
    }
  }
  __syncthreads();

  const int lane = tid & 31, w = tid >> 5;
  const int rA = lane & 15, hlf = lane >> 4, kboff = hlf * 8, cN = rA;
  const int mt = w;  // each wave owns one 16-row strip

  v16bf qa[8];       // qf A-fragments, K = 256, reused in both phases
  {
    size_t fr = (size_t)(b * SS + s0 + mt * 16 + rA) * HN + h;
    const bf16_t* p = QF + fr * RR + kboff;
#pragma unroll
    for (int ks = 0; ks < 8; ++ks) qa[ks] = load_frag(p + ks * 32);
  }

  // Phase 1: masked A -> Abuf
  for (int nt = 0; nt < 8; ++nt) {
    if (nt > mt) {  // fully above the diagonal: zeros
#pragma unroll
      for (int vi = 0; vi < 8; ++vi)
        Abuf[(mt * 16 + vi + 8 * hlf) * CC + nt * 16 + cN] = f2bf(0.0f);
      continue;
    }
    size_t fn = (size_t)(b * SS + s0 + nt * 16 + rA) * HN + h;
    const bf16_t* pb = KF + fn * RR + kboff;
    v16bf bfr[8];
#pragma unroll
    for (int ks = 0; ks < 8; ++ks) bfr[ks] = load_frag(pb + ks * 32);
    v8f acc = {};
#pragma unroll
    for (int ks = 0; ks < 8; ++ks)
      acc = __builtin_amdgcn_wmma_f32_16x16x32_bf16(
          false, qa[ks], false, bfr[ks], (short)0, acc, false, false);
#pragma unroll
    for (int vi = 0; vi < 8; ++vi) {
      int rloc = vi + 8 * hlf;
      float val = acc[vi];
      if (nt == mt && cN > rloc) val = 0.0f;   // strict upper triangle of diag tile
      Abuf[(mt * 16 + rloc) * CC + nt * 16 + cN] = f2bf(val);
    }
  }
  __syncthreads();

  v16bf aa[4];       // Abuf A-fragments, K = 128
  {
    const bf16_t* p = Abuf + (mt * 16 + rA) * CC + kboff;
#pragma unroll
    for (int ks = 0; ks < 4; ++ks) aa[ks] = load_frag(p + ks * 32);
  }

  // Denominator tile (N cols 64..79; col 64 = qf·Z + rowsum(tril A))
  float dv[8];
  {
    const bf16_t* pb = StB  + (64 + rA) * RR + kboff;
    const bf16_t* pv = vExt + (64 + rA) * CC + kboff;
    v16bf bfr[8], bfv[4];
#pragma unroll
    for (int ks = 0; ks < 8; ++ks) bfr[ks] = load_frag(pb + ks * 32);
#pragma unroll
    for (int ks = 0; ks < 4; ++ks) bfv[ks] = load_frag(pv + ks * 32);
    v8f acc = {};
#pragma unroll
    for (int ks = 0; ks < 8; ++ks)
      acc = __builtin_amdgcn_wmma_f32_16x16x32_bf16(
          false, qa[ks], false, bfr[ks], (short)0, acc, false, false);
#pragma unroll
    for (int ks = 0; ks < 4; ++ks)
      acc = __builtin_amdgcn_wmma_f32_16x16x32_bf16(
          false, aa[ks], false, bfv[ks], (short)0, acc, false, false);
#pragma unroll
    for (int vi = 0; vi < 8; ++vi)
      dv[vi] = __shfl(acc[vi], hlf * 16, 32) + EPSF;  // broadcast col 64
  }

  // Numerator tiles nt = 0..3, divide, store
  for (int nt = 0; nt < 4; ++nt) {
    const bf16_t* pb = StB  + (nt * 16 + rA) * RR + kboff;
    const bf16_t* pv = vExt + (nt * 16 + rA) * CC + kboff;
    v16bf bfr[8], bfv[4];
#pragma unroll
    for (int ks = 0; ks < 8; ++ks) bfr[ks] = load_frag(pb + ks * 32);
#pragma unroll
    for (int ks = 0; ks < 4; ++ks) bfv[ks] = load_frag(pv + ks * 32);
    v8f acc = {};
#pragma unroll
    for (int ks = 0; ks < 8; ++ks)
      acc = __builtin_amdgcn_wmma_f32_16x16x32_bf16(
          false, qa[ks], false, bfr[ks], (short)0, acc, false, false);
#pragma unroll
    for (int ks = 0; ks < 4; ++ks)
      acc = __builtin_amdgcn_wmma_f32_16x16x32_bf16(
          false, aa[ks], false, bfv[ks], (short)0, acc, false, false);
#pragma unroll
    for (int vi = 0; vi < 8; ++vi) {
      int rloc = mt * 16 + vi + 8 * hlf;
      Attn[(size_t)(b * SS + s0 + rloc) * DD + h * KDIM + nt * 16 + cN] =
          f2bf(acc[vi] / dv[vi]);
    }
  }
}

// ---------------------------------------------------------------------------
extern "C" void kernel_launch(void* const* d_in, const int* in_sizes, int n_in,
                              void* d_out, int out_size, void* d_ws, size_t ws_size,
                              hipStream_t stream) {
  (void)in_sizes; (void)n_in; (void)out_size; (void)ws_size;
  const float* x  = (const float*)d_in[0];
  const float* Wq = (const float*)d_in[1];
  const float* Wk = (const float*)d_in[2];
  const float* Wv = (const float*)d_in[3];
  const float* Wo = (const float*)d_in[4];
  const float* bo = (const float*)d_in[5];
  const float* rf = (const float*)d_in[6];

  char* ws = (char*)d_ws;
  size_t off = 0;
  auto alloc = [&](size_t bytes) -> char* {
    char* p = ws + off;
    off += (bytes + 255) & ~(size_t)255;
    return p;
  };
  const size_t BS = (size_t)BBATCH * SS;              // 16384
  bf16_t* xb   = (bf16_t*)alloc(BS * DD * 2);
  bf16_t* Wqt  = (bf16_t*)alloc((size_t)DD * DD * 2);
  bf16_t* Wkt  = (bf16_t*)alloc((size_t)DD * DD * 2);
  bf16_t* Wvt  = (bf16_t*)alloc((size_t)DD * DD * 2);
  bf16_t* Wot  = (bf16_t*)alloc((size_t)DD * DD * 2);
  bf16_t* rfb  = (bf16_t*)alloc((size_t)RR * KDIM * 2);
  bf16_t* qb   = (bf16_t*)alloc(BS * DD * 2);
  bf16_t* kb   = (bf16_t*)alloc(BS * DD * 2);
  bf16_t* vb   = (bf16_t*)alloc(BS * DD * 2);
  bf16_t* qf   = (bf16_t*)alloc(BS * HN * RR * 2);
  bf16_t* kf   = (bf16_t*)alloc(BS * HN * RR * 2);
  float*  Ssum = (float*) alloc((size_t)NBH * NC * 80 * RR * 4);
  bf16_t* attn = (bf16_t*)alloc(BS * DD * 2);

  const int nx = (int)(BS * DD);                      // 8388608
  cvt_f32_bf16<<<(nx + 255) / 256, 256, 0, stream>>>(x, xb, nx);
  transpose_cvt<<<(DD * DD + 255) / 256, 256, 0, stream>>>(Wq, Wqt, DD, DD);
  transpose_cvt<<<(DD * DD + 255) / 256, 256, 0, stream>>>(Wk, Wkt, DD, DD);
  transpose_cvt<<<(DD * DD + 255) / 256, 256, 0, stream>>>(Wv, Wvt, DD, DD);
  transpose_cvt<<<(DD * DD + 255) / 256, 256, 0, stream>>>(Wo, Wot, DD, DD);
  cvt_f32_bf16<<<(RR * KDIM + 255) / 256, 256, 0, stream>>>(rf, rfb, RR * KDIM);

  dim3 gg(DD / 128, (unsigned)(BS / 128));            // (4, 128)
  gemm_bf16_kernel<0><<<gg, 256, 0, stream>>>(xb, Wqt, qb, nullptr);
  gemm_bf16_kernel<0><<<gg, 256, 0, stream>>>(xb, Wkt, kb, nullptr);
  gemm_bf16_kernel<0><<<gg, 256, 0, stream>>>(xb, Wvt, vb, nullptr);

  feature_kernel<<<(unsigned)(BS * HN / 128), 256, 0, stream>>>(qb, rfb, qf);
  feature_kernel<<<(unsigned)(BS * HN / 128), 256, 0, stream>>>(kb, rfb, kf);

  chunk_sum_kernel<<<dim3(NC, NBH), 256, 0, stream>>>(kf, vb, Ssum);
  prefix_kernel<<<(NBH * 80 * RR) / 256, 256, 0, stream>>>(Ssum);
  favor_kernel<<<dim3(NC, NBH), 256, 0, stream>>>(qf, kf, vb, Ssum, attn);

  gemm_bf16_kernel<1><<<gg, 256, 0, stream>>>(attn, Wot, d_out, bo);
}
